// RG_LRU_3547642987360
// MI455X (gfx1250) — matrix operand order repaired and checked by reading, not compiled, not run on hardware
//
#include <hip/hip_runtime.h>
#include <hip/hip_bf16.h>
#include <math.h>

// ---------------------------------------------------------------------------
// RG-LRU (Griffin) fused pipeline for MI455X (gfx1250, wave32, WMMA).
//   Pass 0: fp32 -> bf16 conversion of x, W_in, W_gate; precompute -C*softplus(lambda)
//   Pass 1: dual fused GEMM (input_gate & recurrence_gate share the A tile),
//           v_wmma_f32_16x16x32_bf16, double-buffered LDS (1 barrier/iter),
//           fused elementwise epilogue -> alpha, xbeta (fp32)
//   Pass 2-4: chunked linear scan h_t = a_t h_{t-1} + xb_t over T
// Workspace requirement: ~338 MB.
// ---------------------------------------------------------------------------

typedef __bf16 bf16_t;
typedef __attribute__((ext_vector_type(16))) __bf16 bf16x16;
typedef __attribute__((ext_vector_type(8)))  __bf16 bf16x8;
typedef __attribute__((ext_vector_type(4)))  __bf16 bf16x4;
typedef __attribute__((ext_vector_type(8)))  float  f32x8;

#define DIM 2048
#define BB  4
#define TT  4096
#define MM  (BB*TT)        // 16384 rows

// GEMM tiling
#define BM 128
#define BN 128
#define BK 32
#define LDSP 40            // padded LDS row stride in bf16 elems (80 B, 16B-aligned)
#define KITERS (DIM/BK)    // 64

// scan chunking
#define LC 256
#define NC (TT/LC)         // 16

// ------------------------------ pass 0 -------------------------------------

__global__ void cvt_f32_to_bf16(const float* __restrict__ src,
                                bf16_t* __restrict__ dst, int n4) {
  int i = blockIdx.x * blockDim.x + threadIdx.x;
  if (i >= n4) return;
  float4 v = ((const float4*)src)[i];
  bf16x4 o;
  o[0] = (__bf16)v.x; o[1] = (__bf16)v.y; o[2] = (__bf16)v.z; o[3] = (__bf16)v.w;
  ((bf16x4*)dst)[i] = o;
}

__global__ void prep_lambda(const float* __restrict__ lam,
                            float* __restrict__ cLam, int n) {
  int i = blockIdx.x * blockDim.x + threadIdx.x;
  if (i >= n) return;
  // -C * softplus(lambda),  C = 8
  cLam[i] = -8.0f * log1pf(__expf(lam[i]));
}

// ------------------------------ pass 1: fused dual GEMM --------------------
// out[m,n] = sum_k X[m,k] * W[n,k]  (NT GEMM; both operands K-contiguous)
// A-frag per lane (16x32 bf16): lanes 0-15 -> M=lane, K = 0..15 (contiguous)
//                               lanes 16-31 -> M=lane-16, K = 16..31 (contiguous)
// B-frag per lane (32x16 bf16): col = lane%16, K = 16 contiguous starting at
//                               0 (lanes 0-15) or 16 (lanes 16-31)
// C/D 16x16 f32: VGPR i -> M = i + 8*(lane>=16), N = lane%16

struct Stage { uint4 a[2], bi[2], bg[2]; };

__device__ __forceinline__ void stage_load(Stage& s,
                                           const bf16_t* __restrict__ xA,
                                           const bf16_t* __restrict__ Wi,
                                           const bf16_t* __restrict__ Wg,
                                           int mBase, int nBase, int k0, int tid) {
#pragma unroll
  for (int p = 0; p < 2; ++p) {
    const int c  = tid + p * 256;
    const int r  = c >> 2;
    const int c8 = (c & 3) * 8;
    s.a[p]  = *(const uint4*)(&xA[(size_t)(mBase + r) * DIM + k0 + c8]);
    s.bi[p] = *(const uint4*)(&Wi[(size_t)(nBase + r) * DIM + k0 + c8]);
    s.bg[p] = *(const uint4*)(&Wg[(size_t)(nBase + r) * DIM + k0 + c8]);
  }
}

__device__ __forceinline__ void stage_store(const Stage& s,
                                            bf16_t* __restrict__ As,
                                            bf16_t* __restrict__ Bis,
                                            bf16_t* __restrict__ Bgs, int tid) {
#pragma unroll
  for (int p = 0; p < 2; ++p) {
    const int c  = tid + p * 256;
    const int r  = c >> 2;
    const int c8 = (c & 3) * 8;
    *(uint4*)(&As [r * LDSP + c8]) = s.a[p];
    *(uint4*)(&Bis[r * LDSP + c8]) = s.bi[p];
    *(uint4*)(&Bgs[r * LDSP + c8]) = s.bg[p];
  }
}

__global__ __launch_bounds__(256)
void gate_gemm(const bf16_t* __restrict__ xA,
               const bf16_t* __restrict__ Wi,
               const bf16_t* __restrict__ Wg,
               const float*  __restrict__ xf,
               const float*  __restrict__ b_in,
               const float*  __restrict__ b_gate,
               const float*  __restrict__ cLam,
               float* __restrict__ alpha,
               float* __restrict__ xbeta) {
  __shared__ bf16_t As [2][BM * LDSP];
  __shared__ bf16_t Bis[2][BN * LDSP];
  __shared__ bf16_t Bgs[2][BN * LDSP];

  const int tid   = threadIdx.x;
  const int mBase = blockIdx.y * BM;
  const int nBase = blockIdx.x * BN;
  const int wid  = tid >> 5;
  const int lane = tid & 31;
  const int half = lane >> 4;   // 0: lanes 0-15, 1: lanes 16-31
  const int l16  = lane & 15;
  const int wm = wid & 1;       // 2 wave rows  -> 64 M each
  const int wn = wid >> 1;      // 4 wave cols  -> 32 N each

  const f32x8 fz = {0.f, 0.f, 0.f, 0.f, 0.f, 0.f, 0.f, 0.f};
  f32x8 accI[4][2], accG[4][2];
#pragma unroll
  for (int mt = 0; mt < 4; ++mt)
#pragma unroll
    for (int nt = 0; nt < 2; ++nt) { accI[mt][nt] = fz; accG[mt][nt] = fz; }

  union FU { bf16x16 v; bf16x8 h[2]; };

  // prologue: stage tile 0 into buffer 0
  Stage st;
  stage_load(st, xA, Wi, Wg, mBase, nBase, 0, tid);
  stage_store(st, As[0], Bis[0], Bgs[0], tid);
  __syncthreads();

  for (int kt = 0; kt < KITERS; ++kt) {
    const int cur = kt & 1;

    // stage tile k+1 from global into VGPRs (overlaps with WMMA below)
    if (kt + 1 < KITERS)
      stage_load(st, xA, Wi, Wg, mBase, nBase, (kt + 1) * BK, tid);
    // prefetch tile k+2 into cache (global_prefetch_b8)
    if (kt + 2 < KITERS) {
      const int r = tid & 127;
      const int kp = (kt + 2) * BK;
      __builtin_prefetch(&xA[(size_t)(mBase + r) * DIM + kp], 0, 1);
      __builtin_prefetch(&Wi[(size_t)(nBase + r) * DIM + kp], 0, 1);
      __builtin_prefetch(&Wg[(size_t)(nBase + r) * DIM + kp], 0, 1);
    }

    // ---- fragment loads + 16 dual-gate WMMAs per wave per K step ----
    FU bi[2], bg[2];
#pragma unroll
    for (int nt = 0; nt < 2; ++nt) {
      const int col = wn * 32 + nt * 16 + l16;
      const int kb  = half * 16;
      bi[nt].h[0] = *(const bf16x8*)(&Bis[cur][col * LDSP + kb]);
      bi[nt].h[1] = *(const bf16x8*)(&Bis[cur][col * LDSP + kb + 8]);
      bg[nt].h[0] = *(const bf16x8*)(&Bgs[cur][col * LDSP + kb]);
      bg[nt].h[1] = *(const bf16x8*)(&Bgs[cur][col * LDSP + kb + 8]);
    }
#pragma unroll
    for (int mt = 0; mt < 4; ++mt) {
      FU a;
      const int row = wm * 64 + mt * 16 + l16;
      const int kb  = half * 16;
      a.h[0] = *(const bf16x8*)(&As[cur][row * LDSP + kb]);
      a.h[1] = *(const bf16x8*)(&As[cur][row * LDSP + kb + 8]);
#pragma unroll
      for (int nt = 0; nt < 2; ++nt) {
        accI[mt][nt] = __builtin_amdgcn_wmma_f32_16x16x32_bf16(
            false, a.v, false, bi[nt].v, (short)0, accI[mt][nt], false, false);
        accG[mt][nt] = __builtin_amdgcn_wmma_f32_16x16x32_bf16(
            false, a.v, false, bg[nt].v, (short)0, accG[mt][nt], false, false);
      }
    }

    // write tile k+1 into the other buffer; single barrier per iteration
    if (kt + 1 < KITERS)
      stage_store(st, As[cur ^ 1], Bis[cur ^ 1], Bgs[cur ^ 1], tid);
    __syncthreads();
  }

  // ---- fused elementwise epilogue -> alpha, xbeta (fp32) ----
#pragma unroll
  for (int nt = 0; nt < 2; ++nt) {
    const int gn = nBase + wn * 32 + nt * 16 + l16;
    const float bin = b_in[gn];
    const float bgt = b_gate[gn];
    const float cl  = cLam[gn];
#pragma unroll
    for (int mt = 0; mt < 4; ++mt) {
#pragma unroll
      for (int i = 0; i < 8; ++i) {
        const int gm = mBase + wm * 64 + mt * 16 + half * 8 + i;
        const size_t idx = (size_t)gm * DIM + gn;
        const float ig = accI[mt][nt][i] + bin;
        const float rg = accG[mt][nt][i] + bgt;
        const float sr = 1.0f / (1.0f + __expf(-rg));
        const float al = __expf(cl * sr);
        const float be = sqrtf(1.0f - al * al + 1e-6f);
        const float si = 1.0f / (1.0f + __expf(-ig));
        alpha[idx] = al;
        xbeta[idx] = be * si * xf[idx];
      }
    }
  }
}

// ------------------------------ passes 2-4: chunked scan -------------------

__global__ void scan_chunk(const float* __restrict__ alpha,
                           const float* __restrict__ xbeta,
                           float* __restrict__ cA, float* __restrict__ cH) {
  const int d = blockIdx.x * blockDim.x + threadIdx.x;
  const int c = blockIdx.y, b = blockIdx.z;
  const size_t base = ((size_t)b * TT + (size_t)c * LC) * DIM + d;
  float ap = 1.0f, h = 0.0f;
  for (int t = 0; t < LC; ++t) {
    const float a  = alpha[base + (size_t)t * DIM];
    const float xb = xbeta[base + (size_t)t * DIM];
    h = fmaf(a, h, xb);
    ap *= a;
  }
  const size_t ci = ((size_t)b * NC + c) * DIM + d;
  cA[ci] = ap;
  cH[ci] = h;
}

__global__ void scan_carry(const float* __restrict__ cA,
                           const float* __restrict__ cH,
                           float* __restrict__ pH) {
  const int d = blockIdx.x * blockDim.x + threadIdx.x;
  const int b = blockIdx.z;
  float h = 0.0f;
  for (int c = 0; c < NC; ++c) {
    const size_t i = ((size_t)b * NC + c) * DIM + d;
    pH[i] = h;                       // exclusive prefix entering chunk c
    h = fmaf(cA[i], h, cH[i]);
  }
}

__global__ void scan_apply(const float* __restrict__ alpha,
                           const float* __restrict__ xbeta,
                           const float* __restrict__ pH,
                           float* __restrict__ out) {
  const int d = blockIdx.x * blockDim.x + threadIdx.x;
  const int c = blockIdx.y, b = blockIdx.z;
  const size_t base = ((size_t)b * TT + (size_t)c * LC) * DIM + d;
  float h = pH[((size_t)b * NC + c) * DIM + d];
  for (int t = 0; t < LC; ++t) {
    const float a  = alpha[base + (size_t)t * DIM];
    const float xb = xbeta[base + (size_t)t * DIM];
    h = fmaf(a, h, xb);
    out[base + (size_t)t * DIM] = h;
  }
}

// ------------------------------ launcher -----------------------------------

extern "C" void kernel_launch(void* const* d_in, const int* in_sizes, int n_in,
                              void* d_out, int out_size, void* d_ws, size_t ws_size,
                              hipStream_t stream) {
  const float* x      = (const float*)d_in[0];
  const float* W_in   = (const float*)d_in[1];
  const float* b_in   = (const float*)d_in[2];
  const float* W_gate = (const float*)d_in[3];
  const float* b_gate = (const float*)d_in[4];
  const float* lam    = (const float*)d_in[5];
  float* out = (float*)d_out;

  // workspace layout (bytes)
  char* ws = (char*)d_ws;
  const size_t SZ_X16 = (size_t)MM * DIM * 2;        // 67,108,864
  const size_t SZ_W16 = (size_t)DIM * DIM * 2;       //  8,388,608
  const size_t SZ_F32 = (size_t)MM * DIM * 4;        // 134,217,728
  const size_t SZ_CAR = (size_t)BB * NC * DIM * 4;   //     524,288
  bf16_t* xb16 = (bf16_t*)(ws);
  bf16_t* wi16 = (bf16_t*)(ws + SZ_X16);
  bf16_t* wg16 = (bf16_t*)(ws + SZ_X16 + SZ_W16);
  float*  cLam = (float*) (ws + SZ_X16 + 2 * SZ_W16);
  float*  alp  = (float*) (ws + SZ_X16 + 2 * SZ_W16 + 8192);
  float*  xbt  = (float*) (ws + SZ_X16 + 2 * SZ_W16 + 8192 + SZ_F32);
  float*  cA   = (float*) (ws + SZ_X16 + 2 * SZ_W16 + 8192 + 2 * SZ_F32);
  float*  cH   = (float*) (ws + SZ_X16 + 2 * SZ_W16 + 8192 + 2 * SZ_F32 + SZ_CAR);
  float*  pH   = (float*) (ws + SZ_X16 + 2 * SZ_W16 + 8192 + 2 * SZ_F32 + 2 * SZ_CAR);

  // pass 0: conversions
  {
    const int n4x = MM * DIM / 4;   // 8,388,608
    const int n4w = DIM * DIM / 4;  // 1,048,576
    cvt_f32_to_bf16<<<n4x / 256, 256, 0, stream>>>(x,      xb16, n4x);
    cvt_f32_to_bf16<<<n4w / 256, 256, 0, stream>>>(W_in,   wi16, n4w);
    cvt_f32_to_bf16<<<n4w / 256, 256, 0, stream>>>(W_gate, wg16, n4w);
    prep_lambda<<<DIM / 256, 256, 0, stream>>>(lam, cLam, DIM);
  }

  // pass 1: fused dual GEMM + gating
  {
    dim3 grid(DIM / BN, MM / BM, 1);   // 16 x 128
    gate_gemm<<<grid, 256, 0, stream>>>(xb16, wi16, wg16, x, b_in, b_gate,
                                        cLam, alp, xbt);
  }

  // passes 2-4: chunked linear scan over T
  {
    dim3 gA(DIM / 256, NC, BB);
    scan_chunk<<<gA, 256, 0, stream>>>(alp, xbt, cA, cH);
    dim3 gB(DIM / 256, 1, BB);
    scan_carry<<<gB, 256, 0, stream>>>(cA, cH, pH);
    scan_apply<<<gA, 256, 0, stream>>>(alp, xbt, pH, out);
  }
}
